// DCGLC_35845797052729
// MI455X (gfx1250) — matrix operand order repaired and checked by reading, not compiled, not run on hardware
//
#include <hip/hip_runtime.h>
#include <hip/hip_bf16.h>

// ---------------------------------------------------------------------------
// Types for CDNA5 WMMA bf16 path
// ---------------------------------------------------------------------------
typedef __attribute__((ext_vector_type(16))) __bf16 v16bf;
typedef __attribute__((ext_vector_type(8)))  __bf16 v8bf;
typedef __attribute__((ext_vector_type(8)))  float  v8f;

union BfFrag {   // 16 bf16 = one A or B operand of v_wmma_f32_16x16x32_bf16
    v16bf v;
    v8bf  h[2];
};

__device__ __forceinline__ unsigned short f2bf(float f) {
    union { float f; unsigned u; } x; x.f = f;
    unsigned r = x.u + 0x7FFFu + ((x.u >> 16) & 1u);   // round-to-nearest-even
    return (unsigned short)(r >> 16);
}
__device__ __forceinline__ float bf2f(unsigned short h) {
    union { unsigned u; float f; } x; x.u = ((unsigned)h) << 16;
    return x.f;
}

__device__ __forceinline__ void store_out(float* p, float v) { *p = v; }
__device__ __forceinline__ void store_out(unsigned short* p, float v) { *p = f2bf(v); }

// ---------------------------------------------------------------------------
// CDNA5 async global->LDS copy (no VGPR data path, tracked by ASYNCcnt)
// ---------------------------------------------------------------------------
typedef __attribute__((address_space(3))) unsigned short lds_us;

__device__ __forceinline__ unsigned lds_byte_addr(unsigned short* p) {
    return (unsigned)(unsigned long long)(lds_us*)p;   // LDS byte offset
}
__device__ __forceinline__ void async_copy_b128(unsigned lds_addr,
                                                const unsigned short* gptr) {
    asm volatile("global_load_async_to_lds_b128 %0, %1, off"
                 :: "v"(lds_addr), "v"(gptr) : "memory");
}
__device__ __forceinline__ void wait_async0() {
    asm volatile("s_wait_asynccnt 0x0" ::: "memory");
}

// ---------------------------------------------------------------------------
// Elementwise fp32 -> bf16 (vectorized x4)
// ---------------------------------------------------------------------------
__global__ __launch_bounds__(256)
void cvt_f32_bf16(const float* __restrict__ in, unsigned short* __restrict__ out,
                  long long n4)
{
    long long i = (long long)blockIdx.x * 256 + threadIdx.x;
    if (i >= n4) return;
    float4 v = ((const float4*)in)[i];
    ushort4 o;
    o.x = f2bf(v.x); o.y = f2bf(v.y); o.z = f2bf(v.z); o.w = f2bf(v.w);
    ((ushort4*)out)[i] = o;
}

// ---------------------------------------------------------------------------
// Weight transpose + convert: in fp32 [K][N] -> out bf16 [N][K]
// ---------------------------------------------------------------------------
__global__ __launch_bounds__(256)
void transpose_cvt(const float* __restrict__ in, unsigned short* __restrict__ out,
                   int K, int N)
{
    __shared__ float t[32][33];
    const int bn = blockIdx.x * 32;            // N dim
    const int bk = blockIdx.y * 32;            // K dim
    const int tx = threadIdx.x, ty = threadIdx.y;   // 32 x 8
    for (int j = ty; j < 32; j += 8)
        t[j][tx] = in[(size_t)(bk + j) * N + bn + tx];
    __syncthreads();
    for (int j = ty; j < 32; j += 8)
        out[(size_t)(bn + j) * K + bk + tx] = f2bf(t[tx][j]);
}

// ---------------------------------------------------------------------------
// GEMM: Cout[M,Nd] = act(A[M,Kd] @ W[Kd,Nd] + bias) (+ resid)
//   A  : bf16 row-major [M][Kd]
//   Bt : bf16 TRANSPOSED weights [Nd][Kd]  (pre-converted)
//   ACT: 0 = none, 1 = ReLU, 2 = LeakyReLU(0.01) (compile-time)
//   RESID/BIAS: compile-time presence flags -> straight-line epilogue
// Tile: BM=256, BN=128, BK=32. 256 threads = 8 waves (4x2), each wave 64x64
// = 4x4 WMMA 16x16 accumulators (16 v_wmma per K-step per wave).
// Staging: GLOBAL_LOAD_ASYNC_TO_LDS_B128 double-buffered (no staging VGPRs,
// no scratch spills), synced with s_wait_asynccnt + barrier.
// ---------------------------------------------------------------------------
template <typename TOut, int ACT, bool RESID, bool BIAS>
__global__ __launch_bounds__(256)
void gemm_bf16_wmma(const unsigned short* __restrict__ A,
                    const unsigned short* __restrict__ Bt,
                    const float* __restrict__ bias,
                    const unsigned short* __restrict__ resid,
                    TOut* __restrict__ Cout, int M, int Kd, int Nd)
{
    // bf16 tiles, row stride padded to 40 elems (80B: 16B-aligned, low-conflict)
    __shared__ unsigned short lA[2][256 * 40];   // A tile: [row][k]
    __shared__ unsigned short lB[2][128 * 40];   // W tile: [n][k]

    const int tid  = threadIdx.x;
    const int wave = tid >> 5;
    const int lane = tid & 31;
    const int Mbase = blockIdx.x * 256;
    const int Nbase = blockIdx.y * 128;

    v8f acc[4][4];
    v8f zero8;
#pragma unroll
    for (int i = 0; i < 8; ++i) zero8[i] = 0.0f;
#pragma unroll
    for (int mt = 0; mt < 4; ++mt)
#pragma unroll
        for (int nt = 0; nt < 4; ++nt) acc[mt][nt] = zero8;

    const int KT = Kd >> 5;

    // Per-thread staging geometry (16-byte chunks of 8 bf16):
    // A: 256x32 -> 1024 chunks -> 4/thread; row=idx>>2, c8=(idx&3)*8
    // B: 128x32 ->  512 chunks -> 2/thread; n  =idx>>2, k8=(idx&3)*8
    auto stage = [&](int kt, int buf) {
        const unsigned short* Ak = A  + (size_t)kt * 32;
        const unsigned short* Bk = Bt + (size_t)kt * 32;
#pragma unroll
        for (int i = 0; i < 4; ++i) {
            int idx = tid + i * 256;
            async_copy_b128(lds_byte_addr(&lA[buf][(idx >> 2) * 40 + (idx & 3) * 8]),
                            Ak + (size_t)(Mbase + (idx >> 2)) * Kd + (idx & 3) * 8);
        }
#pragma unroll
        for (int i = 0; i < 2; ++i) {
            int idx = tid + i * 256;
            async_copy_b128(lds_byte_addr(&lB[buf][(idx >> 2) * 40 + (idx & 3) * 8]),
                            Bk + (size_t)(Nbase + (idx >> 2)) * Kd + (idx & 3) * 8);
        }
    };

    stage(0, 0);
    wait_async0();
    __syncthreads();

    // Fragment base addresses (ISA 16-bit A / B WMMA layouts)
    const int arow0 = (wave & 3) * 64 + (lane & 15);  // lanes 0-15 carry M rows
    const int akb   = (lane >> 4) * 8;                // K 0-7/16-23 vs 8-15/24-31
    const int bn0   = (wave >> 2) * 64 + (lane & 15); // lanes 0-15 carry N cols
    const int bkb   = (lane >> 4) * 8;                // K 0-7/16-23 vs 8-15/24-31

    for (int kt = 0; kt < KT; ++kt) {
        const int cur = kt & 1, nxt = cur ^ 1;
        const bool more = (kt + 1) < KT;

        if (more) stage(kt + 1, nxt);   // async DMA into the other buffer

        // ---- fragment loads from LDS + 16 WMMAs ----
        BfFrag af[4], bf[4];
#pragma unroll
        for (int mt = 0; mt < 4; ++mt) {
            int ro = (arow0 + mt * 16) * 40 + akb;
            af[mt].h[0] = *(const v8bf*)&lA[cur][ro];
            af[mt].h[1] = *(const v8bf*)&lA[cur][ro + 16];
        }
#pragma unroll
        for (int nt = 0; nt < 4; ++nt) {
            int bo = (bn0 + nt * 16) * 40 + bkb;
            bf[nt].h[0] = *(const v8bf*)&lB[cur][bo];
            bf[nt].h[1] = *(const v8bf*)&lB[cur][bo + 16];
        }
#pragma unroll
        for (int mt = 0; mt < 4; ++mt)
#pragma unroll
            for (int nt = 0; nt < 4; ++nt)
                acc[mt][nt] = __builtin_amdgcn_wmma_f32_16x16x32_bf16(
                    false, af[mt].v, false, bf[nt].v,
                    (short)0, acc[mt][nt], false, false);

        if (more) wait_async0();        // my async writes to nxt are done
        __syncthreads();                // everyone's are, and cur reads done
    }

    // ---- epilogue: bias + activation + optional bf16 residual (branchless) ----
    const int rowW = Mbase + (wave & 3) * 64;
    const int colW = Nbase + (wave >> 2) * 64;
#pragma unroll
    for (int mt = 0; mt < 4; ++mt) {
#pragma unroll
        for (int nt = 0; nt < 4; ++nt) {
            int col  = colW + nt * 16 + (lane & 15);
            int row0 = rowW + mt * 16 + (lane >> 4) * 8;
            float bcol = BIAS ? bias[col] : 0.0f;
#pragma unroll
            for (int r = 0; r < 8; ++r) {
                float v = acc[mt][nt][r] + bcol;
                if (ACT == 1)      v = v > 0.0f ? v : 0.0f;
                else if (ACT == 2) v = v > 0.0f ? v : 0.01f * v;
                size_t o = (size_t)(row0 + r) * Nd + col;
                if (RESID) v += bf2f(resid[o]);
                store_out(&Cout[o], v);
            }
        }
    }
}

// ---------------------------------------------------------------------------
// q/s kernel: one wave per row of z (C=512, K=16, DSUB=32)
// ---------------------------------------------------------------------------
__global__ __launch_bounds__(256)
void qs_kernel(const float* __restrict__ z, const float* __restrict__ zD,
               const float* __restrict__ mu,
               float* __restrict__ qout, float* __restrict__ sout)
{
    __shared__ float smu[16][513];
    __shared__ float munorm[16];
    __shared__ float zrow[8][512];

    const int tid = threadIdx.x;
    for (int i = tid; i < 16 * 512; i += 256) smu[i >> 9][i & 511] = mu[i];
    __syncthreads();
    if (tid < 16) {
        float s = 0.0f;
        for (int c = 0; c < 512; ++c) { float m = smu[tid][c]; s += m * m; }
        munorm[tid] = s;
    }
    __syncthreads();

    const int wave = tid >> 5, lane = tid & 31;
    const int row  = blockIdx.x * 8 + wave;

    const float* zp = z + (size_t)row * 512;
    float zz = 0.0f;
    for (int j = lane; j < 512; j += 32) {
        float v = zp[j];
        zrow[wave][j] = v;
        zz += v * v;
    }
#pragma unroll
    for (int off = 16; off; off >>= 1) zz += __shfl_xor(zz, off, 32);
    __syncthreads();

    const int c = lane & 15, half = lane >> 4;

    float dot = 0.0f;
    const int j0 = half * 256;
    for (int j = j0; j < j0 + 256; ++j) dot += zrow[wave][j] * smu[c][j];
    dot += __shfl_xor(dot, 16, 32);

    float d2 = zz - 2.0f * dot + munorm[c];
    float qv = 1.0f / (1.0f + d2);
    float qs = qv;
#pragma unroll
    for (int off = 8; off; off >>= 1) qs += __shfl_xor(qs, off, 32);
    if (lane < 16) qout[(size_t)row * 16 + lane] = qv / qs;

    const float* zdp = zD + (size_t)row * 512 + c * 32 + half * 16;
    float ss = 0.0f;
#pragma unroll
    for (int j = 0; j < 16; ++j) { float v = zdp[j]; ss += v * v; }
    ss += __shfl_xor(ss, 16, 32);
    float sv = (ss + 32.0f) / 64.0f;
    float st = sv;
#pragma unroll
    for (int off = 8; off; off >>= 1) st += __shfl_xor(st, off, 32);
    if (lane < 16) sout[(size_t)row * 16 + lane] = sv / st;
}

// ---------------------------------------------------------------------------
// Host launcher
// ---------------------------------------------------------------------------
template <typename TOut, int ACT, bool RESID, bool BIAS>
static inline void launch_gemm(const unsigned short* A, const unsigned short* Bt,
                               const float* bias, const unsigned short* resid,
                               TOut* O, int M, int Kd, int Nd, hipStream_t s)
{
    dim3 grid(M / 256, Nd / 128), blk(256);
    gemm_bf16_wmma<TOut, ACT, RESID, BIAS>
        <<<grid, blk, 0, s>>>(A, Bt, bias, resid, O, M, Kd, Nd);
}

extern "C" void kernel_launch(void* const* d_in, const int* in_sizes, int n_in,
                              void* d_out, int out_size, void* d_ws, size_t ws_size,
                              hipStream_t stream)
{
    const int N = 16384, E = 2048, H = 1024, K = 16, C = 512;

    const float* y_in = (const float*)d_in[0];
    const float* pW1 = (const float*)d_in[1],  *pb1 = (const float*)d_in[2];
    const float* pW2 = (const float*)d_in[3],  *pb2 = (const float*)d_in[4];
    const float* fW1 = (const float*)d_in[5],  *fb1 = (const float*)d_in[6];
    const float* fW2 = (const float*)d_in[7],  *fb2 = (const float*)d_in[8];
    const float* fW3 = (const float*)d_in[9],  *fb3 = (const float*)d_in[10];
    const float* fWs = (const float*)d_in[11], *fbs = (const float*)d_in[12];
    const float* cW1 = (const float*)d_in[13], *cb1 = (const float*)d_in[14];
    const float* cW2 = (const float*)d_in[15], *cb2 = (const float*)d_in[16];
    const float* cWs = (const float*)d_in[17], *cbs = (const float*)d_in[18];
    const float* mu  = (const float*)d_in[19];
    const float* Dm  = (const float*)d_in[20];

    float* out   = (float*)d_out;
    float* z_out = out;                                   // [N, C] fp32
    float* q_out = out + (size_t)N * C;                   // [N, K]
    float* s_out = q_out + (size_t)N * K;                 // [N, K]
    float* g_out = s_out + (size_t)N * K;                 // [N, E] fp32

    // ---- workspace carve ----
    char* w = (char*)d_ws;
    auto alloc = [&](size_t bytes) -> char* {
        char* p = w; w += (bytes + 255) & ~(size_t)255; return p;
    };
    unsigned short* a0 = (unsigned short*)alloc((size_t)N * E * 2);  // act ping-pong
    unsigned short* a1 = (unsigned short*)alloc((size_t)N * E * 2);
    unsigned short* a2 = (unsigned short*)alloc((size_t)N * E * 2);
    unsigned short* pW1t = (unsigned short*)alloc((size_t)E * E * 2);
    unsigned short* pW2t = (unsigned short*)alloc((size_t)E * E * 2);
    unsigned short* fW1t = (unsigned short*)alloc((size_t)E * E * 2);
    unsigned short* fW2t = (unsigned short*)alloc((size_t)E * E * 2);
    unsigned short* fW3t = (unsigned short*)alloc((size_t)E * E * 2);
    unsigned short* fWst = (unsigned short*)alloc((size_t)E * E * 2);
    unsigned short* cW1t = (unsigned short*)alloc((size_t)E * H * 2);
    unsigned short* cW2t = (unsigned short*)alloc((size_t)H * C * 2);
    unsigned short* cWst = (unsigned short*)alloc((size_t)E * C * 2);
    unsigned short* Dt   = (unsigned short*)alloc((size_t)C * C * 2);
    float*          zDws = (float*)alloc((size_t)N * C * 4);

    dim3 blk(256);
    auto CVT = [&](const float* in, unsigned short* o, size_t n) {
        long long n4 = (long long)(n / 4);
        cvt_f32_bf16<<<dim3((unsigned)((n4 + 255) / 256)), blk, 0, stream>>>(in, o, n4);
    };
    auto TRC = [&](const float* in, unsigned short* o, int Kd, int Nd) {
        transpose_cvt<<<dim3(Nd / 32, Kd / 32), dim3(32, 8), 0, stream>>>(in, o, Kd, Nd);
    };

    // ---- pre-convert: weights transposed to bf16 [N][K]; y_in to bf16 ----
    TRC(pW1, pW1t, E, E);  TRC(pW2, pW2t, E, E);
    TRC(fW1, fW1t, E, E);  TRC(fW2, fW2t, E, E);
    TRC(fW3, fW3t, E, E);  TRC(fWs, fWst, E, E);
    TRC(cW1, cW1t, E, H);  TRC(cW2, cW2t, H, C);
    TRC(cWs, cWst, E, C);  TRC(Dm,  Dt,   C, C);
    CVT(y_in, a2, (size_t)N * E);

    // ---- GEMM chain (all operands bf16, acc fp32) ----
    launch_gemm<unsigned short, 2, false, true>(a2, pW1t, pb1, nullptr, a0, N, E, E, stream);
    launch_gemm<unsigned short, 0, false, true>(a0, pW2t, pb2, nullptr, a1, N, E, E, stream);
    launch_gemm<unsigned short, 1, false, true>(a1, fW1t, fb1, nullptr, a0, N, E, E, stream);
    launch_gemm<unsigned short, 1, false, true>(a0, fW2t, fb2, nullptr, a2, N, E, E, stream);
    launch_gemm<unsigned short, 1, false, true>(a2, fW3t, fb3, nullptr, a0, N, E, E, stream);
    launch_gemm<float,          0, true,  true>(a1, fWst, fbs, a0, g_out, N, E, E, stream);
    CVT(g_out, a1, (size_t)N * E);                               // g -> bf16
    launch_gemm<unsigned short, 2, false, true>(a1, cW1t, cb1, nullptr, a0, N, E, H, stream);
    launch_gemm<unsigned short, 2, false, true>(a0, cW2t, cb2, nullptr, a2, N, H, C, stream);
    launch_gemm<float,          0, true,  true>(a1, cWst, cbs, a2, z_out, N, E, C, stream);
    CVT(z_out, a0, (size_t)N * C);                               // z -> bf16
    launch_gemm<float,          0, false, false>(a0, Dt, nullptr, nullptr, zDws, N, C, C, stream);
    // ---- DEC q + subspace s ----
    qs_kernel<<<dim3(N / 8), blk, 0, stream>>>(z_out, zDws, mu, q_out, s_out);
}
